// MultiHeadCrossAttention_5738076308176
// MI455X (gfx1250) — matrix operand order repaired and checked by reading, not compiled, not run on hardware
//
#include <hip/hip_runtime.h>
#include <hip/hip_bf16.h>

typedef __bf16 bf16_t;
typedef __attribute__((ext_vector_type(16))) __bf16 v16bf;
typedef __attribute__((ext_vector_type(8)))  __bf16 v8bf;
typedef __attribute__((ext_vector_type(8)))  float  v8f;
typedef __attribute__((ext_vector_type(4)))  unsigned tdm_v4u;
typedef __attribute__((ext_vector_type(8)))  int      tdm_v8i;
typedef __attribute__((ext_vector_type(4)))  int      tdm_v4i;

namespace {
constexpr int kB = 8, kN = 512, kH = 12, kD = 128;
constexpr int kInner = kH * kD;           // 1536
constexpr int kDrugDim = 768;
constexpr int kTargetDim = 2560;
constexpr float kScale = 0.08838834764831845f; // 128^-0.5
constexpr float kNeg = -1000000.0f;
}

// ---------------------------------------------------------------------------
// bf16 packing helpers (lower to v_cvt_pk_bf16_f32 / pack ops)
// ---------------------------------------------------------------------------
__device__ __forceinline__ unsigned pack_bf16(float a, float b) {
  union { bf16_t h[2]; unsigned u; } p;
  p.h[0] = (bf16_t)a; p.h[1] = (bf16_t)b;
  return p.u;
}
__device__ __forceinline__ unsigned pack2_bf16(bf16_t a, bf16_t b) {
  union { bf16_t h[2]; unsigned u; } p;
  p.h[0] = a; p.h[1] = b;
  return p.u;
}

// ---------------------------------------------------------------------------
// TDM: 2-D tile DMA global -> LDS (bf16 elements). Issued once per wave;
// caller gates to a single wave and fences with s_wait_tensorcnt.
// D# layout per CDNA5 ISA ch.8: group0 = count/lds_addr/global_addr/type,
// group1 = data_size + tensor dims 128-wide rows + tile dims + row stride.
// ---------------------------------------------------------------------------
__device__ __forceinline__ void tdm_load_tile_2d(const bf16_t* gsrc, bf16_t* ldst,
                                                 int rows, int cols, int rstride) {
  const unsigned long long ga = (unsigned long long)(uintptr_t)gsrc;
  const unsigned lds = (unsigned)(uintptr_t)ldst;
  tdm_v4u g0;
  g0[0] = 1u;                                             // count = 1 (valid)
  g0[1] = lds;                                            // lds_addr (bytes)
  g0[2] = (unsigned)(ga & 0xffffffffu);                   // global_addr lo
  g0[3] = (unsigned)((ga >> 32) & 0x1ffffffu) | (2u << 30); // addr hi | type=2
  tdm_v8i g1;
  g1[0] = 1 << 16;                                        // data_size = 2 bytes
  g1[1] = (cols & 0xffff) << 16;                          // tensor_dim0 lo
  g1[2] = ((unsigned)cols >> 16) | ((rows & 0xffff) << 16);   // td0 hi | td1 lo
  g1[3] = ((unsigned)rows >> 16) | ((cols & 0xffff) << 16);   // td1 hi | tile_dim0
  g1[4] = rows & 0xffff;                                  // tile_dim1 (dim2=0)
  g1[5] = rstride;                                        // tensor_dim0_stride
  g1[6] = 0;
  g1[7] = 0;
  tdm_v4i gz = {0, 0, 0, 0};
  tdm_v8i gz8 = {0, 0, 0, 0, 0, 0, 0, 0};
  __builtin_amdgcn_tensor_load_to_lds(g0, g1, gz, gz, gz8, 0);
}

// ---------------------------------------------------------------------------
// Fragment loaders (wave32, 16x16x32 bf16 WMMA).
// A (16x32, MxK): lane L holds row r=L&15; K = half*8+{0..7}, 16+half*8+{0..7}
// B (32x16, KxN): LDS holds B^T as [n][k]; lane L holds col n=L&15;
//                 K = half*8+{0..7}, 16+half*8+{0..7} (same as A).
// ---------------------------------------------------------------------------
__device__ __forceinline__ v16bf load_frag_a(const bf16_t* base, int ldRow, int lane) {
  const int r = lane & 15, h = lane >> 4;
  const bf16_t* p = base + r * ldRow + h * 8;
  v8bf lo = *(const v8bf*)(p);
  v8bf hi = *(const v8bf*)(p + 16);
  v16bf f;
#pragma unroll
  for (int i = 0; i < 8; ++i) { f[i] = lo[i]; f[i + 8] = hi[i]; }
  return f;
}

__device__ __forceinline__ v16bf load_frag_b(const bf16_t* baseT, int ldRow, int lane) {
  const int n = lane & 15, h = lane >> 4;
  const bf16_t* p = baseT + n * ldRow + h * 8;
  v8bf lo = *(const v8bf*)(p);
  v8bf hi = *(const v8bf*)(p + 16);
  v16bf f;
#pragma unroll
  for (int i = 0; i < 8; ++i) { f[i] = lo[i]; f[i + 8] = hi[i]; }
  return f;
}

__device__ __forceinline__ v8f wmma_bf16(v16bf a, v16bf b, v8f c) {
  return __builtin_amdgcn_wmma_f32_16x16x32_bf16(
      false, a, false, b, (short)0, c, false, false);
}

// ---------------------------------------------------------------------------
// Tile staging: f32 global -> bf16 LDS, fully packed stores.
// ---------------------------------------------------------------------------
__device__ __forceinline__ void stage_a_tile(bf16_t* As, const float* Ab,
                                             int m0, int K, int k0, int tid) {
  const int r = tid >> 2, c = (tid & 3) * 8;
  const float4* s = (const float4*)(Ab + (size_t)(m0 + r) * K + (k0 + c));
  float4 x = s[0], y = s[1];
  uint4 w;
  w.x = pack_bf16(x.x, x.y); w.y = pack_bf16(x.z, x.w);
  w.z = pack_bf16(y.x, y.y); w.w = pack_bf16(y.z, y.w);
  *(uint4*)(As + r * 32 + c) = w;
}

__device__ __forceinline__ void stage_b_tile(bf16_t* Bt, const float* W,
                                             int n0, int N, int k0, int tid) {
  const int kk = (tid >> 4) * 2, nn = (tid & 15) * 8;
  const float* r0 = W + (size_t)(k0 + kk) * N + (n0 + nn);
  const float* r1 = r0 + N;
  float4 a0 = ((const float4*)r0)[0], a1 = ((const float4*)r0)[1];
  float4 b0 = ((const float4*)r1)[0], b1 = ((const float4*)r1)[1];
  const float ea[8] = {a0.x, a0.y, a0.z, a0.w, a1.x, a1.y, a1.z, a1.w};
  const float eb[8] = {b0.x, b0.y, b0.z, b0.w, b1.x, b1.y, b1.z, b1.w};
#pragma unroll
  for (int u = 0; u < 8; ++u)
    *(unsigned*)(Bt + (nn + u) * 32 + kk) = pack_bf16(ea[u], eb[u]);
}

// ---------------------------------------------------------------------------
// Shared GEMM core: 64x128 block tile, k-step 32, double-buffered LDS,
// 8 waves (2x4) each computing a 32x32 macro-tile (2x2 WMMA frags).
// ---------------------------------------------------------------------------
struct GemmAcc { v8f a[2][2]; };

__device__ __forceinline__ void gemm_core(GemmAcc& G, bf16_t* AsBuf, bf16_t* BtBuf,
                                          const float* Ab, const float* W,
                                          int m0, int n0, int K, int N,
                                          int tid, int lane, int wm, int wn) {
#pragma unroll
  for (int i = 0; i < 2; ++i)
#pragma unroll
    for (int j = 0; j < 2; ++j)
#pragma unroll
      for (int v = 0; v < 8; ++v) G.a[i][j][v] = 0.0f;

  stage_a_tile(AsBuf, Ab, m0, K, 0, tid);
  stage_b_tile(BtBuf, W, n0, N, 0, tid);

  int cur = 0;
  for (int k0 = 0; k0 < K; k0 += 32) {
    __syncthreads();
    if (k0 + 32 < K) {
      stage_a_tile(AsBuf + (cur ^ 1) * (64 * 32), Ab, m0, K, k0 + 32, tid);
      stage_b_tile(BtBuf + (cur ^ 1) * (128 * 32), W, n0, N, k0 + 32, tid);
    }
    if (k0 + 64 < K) {  // pull the k+2 tile toward the WGP (global_prefetch_b8)
      __builtin_prefetch(Ab + (size_t)(m0 + (tid >> 2)) * K + k0 + 64, 0, 1);
      __builtin_prefetch(W + (size_t)(k0 + 64 + (tid >> 4)) * N + n0, 0, 1);
    }
    const bf16_t* As = AsBuf + cur * (64 * 32);
    const bf16_t* Bt = BtBuf + cur * (128 * 32);
    v16bf bfr[2];
#pragma unroll
    for (int j = 0; j < 2; ++j)
      bfr[j] = load_frag_b(Bt + (wn * 32 + j * 16) * 32, 32, lane);
#pragma unroll
    for (int i = 0; i < 2; ++i) {
      v16bf afr = load_frag_a(As + (wm * 32 + i * 16) * 32, 32, lane);
#pragma unroll
      for (int j = 0; j < 2; ++j)
        G.a[i][j] = wmma_bf16(afr, bfr[j], G.a[i][j]);
    }
    cur ^= 1;
  }
}

// ---------------------------------------------------------------------------
// Projection GEMM: dst(b,h,m,d) = bf16( A(b,m,:) @ W(:,n) ), n = h*128 + d
// ---------------------------------------------------------------------------
__global__ __launch_bounds__(256) void gemm_qkv_kernel(
    const float* __restrict__ A, const float* __restrict__ W,
    bf16_t* __restrict__ dst, int M, int K, int N)
{
  __shared__ bf16_t AsBuf[2 * 64 * 32];
  __shared__ bf16_t BtBuf[2 * 128 * 32];
  const int tid = threadIdx.x, lane = tid & 31, wave = tid >> 5;
  const int wm = wave >> 2, wn = wave & 3;
  const int n0 = blockIdx.x * 128, m0 = blockIdx.y * 64, b = blockIdx.z;
  const float* Ab = A + (size_t)b * M * K;

  GemmAcc G;
  gemm_core(G, AsBuf, BtBuf, Ab, W, m0, n0, K, N, tid, lane, wm, wn);

  const int half = lane >> 4, nl = lane & 15;
#pragma unroll
  for (int i = 0; i < 2; ++i)
#pragma unroll
    for (int j = 0; j < 2; ++j)
#pragma unroll
      for (int v = 0; v < 8; ++v) {
        const int m = m0 + wm * 32 + i * 16 + v + 8 * half;
        const int nc = n0 + wn * 32 + j * 16 + nl;
        const int h = nc >> 7, d = nc & 127;
        dst[(((size_t)b * kH + h) * M + m) * kD + d] = (bf16_t)G.a[i][j][v];
      }
}

// ---------------------------------------------------------------------------
// Output GEMM: out(b,m,n) = ctx(b,m,:) @ Wo(:,n) + bo(n) + res(b,m,n)
// ---------------------------------------------------------------------------
__global__ __launch_bounds__(256) void gemm_out_kernel(
    const float* __restrict__ Ctx, const float* __restrict__ Wo,
    const float* __restrict__ bo, const float* __restrict__ Res,
    float* __restrict__ Out, int M, int K, int N)
{
  __shared__ bf16_t AsBuf[2 * 64 * 32];
  __shared__ bf16_t BtBuf[2 * 128 * 32];
  const int tid = threadIdx.x, lane = tid & 31, wave = tid >> 5;
  const int wm = wave >> 2, wn = wave & 3;
  const int n0 = blockIdx.x * 128, m0 = blockIdx.y * 64, b = blockIdx.z;
  const float* Ab = Ctx + (size_t)b * M * K;

  GemmAcc G;
  gemm_core(G, AsBuf, BtBuf, Ab, Wo, m0, n0, K, N, tid, lane, wm, wn);

  const int half = lane >> 4, nl = lane & 15;
#pragma unroll
  for (int i = 0; i < 2; ++i)
#pragma unroll
    for (int j = 0; j < 2; ++j)
#pragma unroll
      for (int v = 0; v < 8; ++v) {
        const int m = m0 + wm * 32 + i * 16 + v + 8 * half;
        const int nc = n0 + wn * 32 + j * 16 + nl;
        const size_t off = ((size_t)b * M + m) * N + nc;
        Out[off] = G.a[i][j][v] + bo[nc] + Res[off];
      }
}

// ---------------------------------------------------------------------------
// Flash attention: grid (N/128, B*H). 8 waves x 16 query rows; online softmax
// over 32-key tiles. K tiles arrive via TDM DMA (wave 0, TENSORcnt-fenced),
// V^T tiles staged manually (transpose), both double-buffered over Q arena.
// ---------------------------------------------------------------------------
__device__ __forceinline__ void stage_vt(bf16_t* Vt, const bf16_t* Vg,
                                         int kt, int tid) {
  // V^T tile [d][key]: k-pair x 8 d per thread, packed b32 stores
  const int kk = (tid >> 4) * 2, dd = (tid & 15) * 8;
  const v8bf r0 = *(const v8bf*)(Vg + (size_t)(kt + kk) * kD + dd);
  const v8bf r1 = *(const v8bf*)(Vg + (size_t)(kt + kk + 1) * kD + dd);
#pragma unroll
  for (int u = 0; u < 8; ++u)
    *(unsigned*)(Vt + (dd + u) * 32 + kk) = pack2_bf16(r0[u], r1[u]);
}

__global__ __launch_bounds__(256) void attn_kernel(
    const bf16_t* __restrict__ Qb, const bf16_t* __restrict__ Kb,
    const bf16_t* __restrict__ Vb, const int* __restrict__ qmask,
    const int* __restrict__ kmask, float* __restrict__ Ctx)
{
  // 40KB arena (20480 bf16): [0,16384) Q staging in the prologue, later
  // aliased as Ks ping-pong [0,4096)+[4096,8192) and Vt ping-pong
  // [8192,12288)+[12288,16384). P scratch lives at [16384,20480).
  __shared__ __align__(16) bf16_t smem[20480];

  const int tid = threadIdx.x, lane = tid & 31, wave = tid >> 5;
  const int half = lane >> 4, nl = lane & 15;
  const int bh = blockIdx.y;
  const int b = bh / kH, h = bh % kH;
  const int q0 = blockIdx.x * 128;

  const bf16_t* Qg = Qb + (size_t)bh * kN * kD;
  const bf16_t* Kg = Kb + (size_t)bh * kN * kD;
  const bf16_t* Vg = Vb + (size_t)bh * kN * kD;

  { // stage Q once, then hoist all Q fragments to VGPRs
    const v8bf* s = (const v8bf*)(Qg + (size_t)q0 * kD);
    v8bf* d = (v8bf*)smem;
    for (int i = tid; i < 128 * kD / 8; i += 256) d[i] = s[i];
  }
  __syncthreads();
  v16bf qf[4];
#pragma unroll
  for (int c = 0; c < 4; ++c)
    qf[c] = load_frag_a(smem + (wave * 16) * kD + c * 32, kD, lane);
  __syncthreads();  // everyone done reading Q before arena reuse

  int rmask[8];
  float rowmax[8], rowsum[8];
  v8f o[8];
#pragma unroll
  for (int v = 0; v < 8; ++v) {
    rmask[v] = qmask[b * kN + q0 + wave * 16 + 8 * half + v];
    rowmax[v] = -1e30f;
    rowsum[v] = 0.0f;
  }
#pragma unroll
  for (int dj = 0; dj < 8; ++dj)
#pragma unroll
    for (int v = 0; v < 8; ++v) o[dj][v] = 0.0f;

  bf16_t* Ps = smem + 16384;  // per-wave P scratch (A layout), 8KB
  if (wave == 0) tdm_load_tile_2d(Kg, smem, /*rows=*/32, /*cols=*/kD, kD);
  stage_vt(smem + 8192, Vg, 0, tid);

  int cur = 0;
  for (int kt = 0; kt < kN; kt += 32) {
    if (wave == 0) __builtin_amdgcn_s_wait_tensorcnt(0);  // K tile (cur) landed
    __syncthreads();  // publish K/V for cur; nxt buffers free for overwrite
    if (kt + 32 < kN) {
      if (wave == 0)
        tdm_load_tile_2d(Kg + (size_t)(kt + 32) * kD,
                         smem + (cur ^ 1) * 4096, 32, kD, kD);
      stage_vt(smem + 8192 + (cur ^ 1) * 4096, Vg, kt + 32, tid);
    }
    if (kt + 64 < kN)
      __builtin_prefetch(Vg + (size_t)(kt + 64) * kD + tid * 16, 0, 1);
    const bf16_t* Ks = smem + cur * 4096;
    const bf16_t* Vt = smem + 8192 + cur * 4096;

    // scores for two 16-key tiles (contraction over d = 4 x 32)
    v8f s0, s1;
#pragma unroll
    for (int v = 0; v < 8; ++v) { s0[v] = 0.0f; s1[v] = 0.0f; }
#pragma unroll
    for (int c = 0; c < 4; ++c) {
      v16bf k0f = load_frag_b(Ks + 0 * kD + c * 32, kD, lane);
      v16bf k1f = load_frag_b(Ks + 16 * kD + c * 32, kD, lane);
      s0 = wmma_bf16(qf[c], k0f, s0);
      s1 = wmma_bf16(qf[c], k1f, s1);
    }
    const int pm0 = kmask[b * kN + kt + nl];
    const int pm1 = kmask[b * kN + kt + 16 + nl];

    float alpha[8];
#pragma unroll
    for (int v = 0; v < 8; ++v) {
      float x0 = s0[v] * kScale;
      float x1 = s1[v] * kScale;
      if (pm0 == 0) x0 = kNeg;
      if (pm1 == 0) x1 = kNeg;
      if (rmask[v] == 0) { x0 = kNeg; x1 = kNeg; }
      float t = fmaxf(x0, x1);
      t = fmaxf(t, __shfl_xor(t, 1, 32));
      t = fmaxf(t, __shfl_xor(t, 2, 32));
      t = fmaxf(t, __shfl_xor(t, 4, 32));
      t = fmaxf(t, __shfl_xor(t, 8, 32));
      const float mnew = fmaxf(rowmax[v], t);
      alpha[v] = __expf(rowmax[v] - mnew);
      rowmax[v] = mnew;
      const float p0 = __expf(x0 - mnew);
      const float p1 = __expf(x1 - mnew);
      float rs = p0 + p1;
      rs += __shfl_xor(rs, 1, 32);
      rs += __shfl_xor(rs, 2, 32);
      rs += __shfl_xor(rs, 4, 32);
      rs += __shfl_xor(rs, 8, 32);
      rowsum[v] = rowsum[v] * alpha[v] + rs;
      // C layout -> A layout bounce through per-wave LDS scratch
      Ps[wave * 512 + (v + 8 * half) * 32 + nl] = (bf16_t)p0;
      Ps[wave * 512 + (v + 8 * half) * 32 + 16 + nl] = (bf16_t)p1;
    }
#pragma unroll
    for (int dj = 0; dj < 8; ++dj)
#pragma unroll
      for (int v = 0; v < 8; ++v) o[dj][v] *= alpha[v];

    // O += P(16x32) @ V(32x128)
    v16bf pf = load_frag_a(Ps + wave * 512, 32, lane);
#pragma unroll
    for (int dj = 0; dj < 8; ++dj) {
      v16bf vf = load_frag_b(Vt + (dj * 16) * 32, 32, lane);
      o[dj] = wmma_bf16(pf, vf, o[dj]);
    }
    cur ^= 1;
  }

#pragma unroll
  for (int dj = 0; dj < 8; ++dj)
#pragma unroll
    for (int v = 0; v < 8; ++v) {
      const int m = q0 + wave * 16 + v + 8 * half;
      const int dd = dj * 16 + nl;
      Ctx[((size_t)b * kN + m) * kInner + h * kD + dd] = o[dj][v] / rowsum[v];
    }
}

// ---------------------------------------------------------------------------
extern "C" void kernel_launch(void* const* d_in, const int* in_sizes, int n_in,
                              void* d_out, int out_size, void* d_ws, size_t ws_size,
                              hipStream_t stream) {
  (void)in_sizes; (void)n_in; (void)out_size; (void)ws_size;
  const float* drug      = (const float*)d_in[0];
  const float* target    = (const float*)d_in[1];
  const int*   drug_mask = (const int*)d_in[2];
  const int*   pro_mask  = (const int*)d_in[3];
  const float* Wq = (const float*)d_in[4];
  const float* Wk = (const float*)d_in[5];
  const float* Wv = (const float*)d_in[6];
  const float* Wo = (const float*)d_in[7];
  const float* bo = (const float*)d_in[8];
  float* out = (float*)d_out;

  const size_t perTensor = (size_t)kB * kH * kN * kD;   // bf16 elems
  bf16_t* qb = (bf16_t*)d_ws;
  bf16_t* kb = qb + perTensor;
  bf16_t* vb = kb + perTensor;
  float*  ctx = (float*)(vb + perTensor);               // (B, N, INNER) f32

  dim3 blk(256, 1, 1);
  gemm_qkv_kernel<<<dim3(kInner / 128, kN / 64, kB), blk, 0, stream>>>(
      drug, Wq, qb, kN, kDrugDim, kInner);
  gemm_qkv_kernel<<<dim3(kInner / 128, kN / 64, kB), blk, 0, stream>>>(
      target, Wk, kb, kN, kTargetDim, kInner);
  gemm_qkv_kernel<<<dim3(kInner / 128, kN / 64, kB), blk, 0, stream>>>(
      target, Wv, vb, kN, kTargetDim, kInner);
  attn_kernel<<<dim3(kN / 128, kB * kH), blk, 0, stream>>>(
      qb, kb, vb, drug_mask, pro_mask, ctx);
  gemm_out_kernel<<<dim3(kTargetDim / 128, kN / 64, kB), blk, 0, stream>>>(
      ctx, Wo, bo, target, out, kN, kInner, kTargetDim);
}
